// Qwen3AttentionTP_57011395887281
// MI455X (gfx1250) — matrix operand order repaired and checked by reading, not compiled, not run on hardware
//
#include <hip/hip_runtime.h>
#include <hip/hip_bf16.h>

// ---------------------------------------------------------------------------
// Qwen3 attention block for MI455X (gfx1250, wave32, WMMA).
// All heavy math runs through v_wmma_f32_16x16x32_bf16 (bf16 in, f32 accum).
// ---------------------------------------------------------------------------

#define B_  2
#define S_  2048
#define D_  2048
#define H_  16
#define HD_ 128
#define MTOT (B_ * S_)          // 4096 rows of tokens

typedef __attribute__((ext_vector_type(16))) __bf16 v16bf;
typedef __attribute__((ext_vector_type(8)))  float  v8f;

__device__ inline v8f wmma_bf16(v16bf a, v16bf b, v8f c) {
  // D = A(16x32 bf16) * B(32x16 bf16) + C(16x16 f32)
  return __builtin_amdgcn_wmma_f32_16x16x32_bf16(
      /*neg_a=*/false, a, /*neg_b=*/false, b,
      /*c_mod=*/(short)0, c, /*reuse_a=*/false, /*reuse_b=*/false);
}

// ---------------------------------------------------------------------------
// f32 -> bf16 conversion
// ---------------------------------------------------------------------------
__global__ void cvt_f32_bf16(const float* __restrict__ in,
                             __bf16* __restrict__ out, long n) {
  long i = (long)blockIdx.x * blockDim.x + threadIdx.x;
  if (i < n) out[i] = (__bf16)in[i];
}

// ---------------------------------------------------------------------------
// WMMA GEMM:  C[M,N] = A[M,K] * Wt[N,K]^T (+ bias)
//   mode 0: write bf16, transposed to [B,H,S,HD]  (projection epilogue)
//   mode 1: write f32 row-major [M,N]             (output projection)
// Block tile 128x64, 8 waves, each wave computes a 32x32 region (2x2 WMMA).
// ---------------------------------------------------------------------------
#define TM 128
#define TN 64
#define TK 32

__global__ __launch_bounds__(256)
void gemm_bf16_wmma(const __bf16* __restrict__ A,
                    const __bf16* __restrict__ Wt,
                    const float* __restrict__ bias,
                    void* __restrict__ Cout,
                    int M, int N, int K, int mode) {
  __shared__ __bf16 As[TM][TK + 8];
  __shared__ __bf16 Bs[TN][TK + 8];

  const int tid  = threadIdx.x;
  const int wid  = tid >> 5;
  const int lane = tid & 31;
  const int half = lane >> 4;
  const int l16  = lane & 15;
  const int m0 = blockIdx.y * TM;
  const int n0 = blockIdx.x * TN;
  const int wm = (wid & 3) * 32;      // wave row offset in tile
  const int wn = (wid >> 2) * 32;     // wave col offset in tile

  v8f acc[2][2] = {};

  for (int k0 = 0; k0 < K; k0 += TK) {
    __syncthreads();
    {   // stage A tile: 128x32 bf16, 16 contiguous elems per thread
      int r = tid >> 1, c = (tid & 1) * 16;
      const __bf16* src = A + (size_t)(m0 + r) * K + k0 + c;
      if (k0 + TK < K) __builtin_prefetch(src + TK, 0, 0);
#pragma unroll
      for (int i = 0; i < 16; ++i) As[r][c + i] = src[i];
    }
    {   // stage W tile: 64x32 bf16, 8 contiguous elems per thread
      int r = tid >> 2, c = (tid & 3) * 8;
      const __bf16* src = Wt + (size_t)(n0 + r) * K + k0 + c;
      if (k0 + TK < K) __builtin_prefetch(src + TK, 0, 0);
#pragma unroll
      for (int i = 0; i < 8; ++i) Bs[r][c + i] = src[i];
    }
    __syncthreads();

    v16bf afr[2], bfr[2];
#pragma unroll
    for (int mt = 0; mt < 2; ++mt) {
      int mr = wm + mt * 16 + l16;            // A layout: lane = M
#pragma unroll
      for (int e = 0; e < 16; ++e) {
        int kk = half * 16 + e;
        afr[mt][e] = As[mr][kk];
      }
    }
#pragma unroll
    for (int nt = 0; nt < 2; ++nt) {
      int nr = wn + nt * 16 + l16;            // B layout: lane = N
#pragma unroll
      for (int e = 0; e < 16; ++e)
        bfr[nt][e] = Bs[nr][half * 16 + e];
    }
#pragma unroll
    for (int mt = 0; mt < 2; ++mt)
#pragma unroll
      for (int nt = 0; nt < 2; ++nt)
        acc[mt][nt] = wmma_bf16(afr[mt], bfr[nt], acc[mt][nt]);
  }

  // epilogue: C element r -> (M = r + 8*half, N = l16)
#pragma unroll
  for (int mt = 0; mt < 2; ++mt)
#pragma unroll
    for (int nt = 0; nt < 2; ++nt)
#pragma unroll
      for (int r = 0; r < 8; ++r) {
        int gm = m0 + wm + mt * 16 + r + 8 * half;
        int gn = n0 + wn + nt * 16 + l16;
        float v = acc[mt][nt][r];
        if (bias) v += bias[gn];
        if (mode == 0) {
          int b = gm >> 11, s = gm & (S_ - 1);
          int h = gn >> 7,  d = gn & (HD_ - 1);
          ((__bf16*)Cout)[(((size_t)b * H_ + h) * S_ + s) * HD_ + d] = (__bf16)v;
        } else {
          ((float*)Cout)[(size_t)gm * N + gn] = v;
        }
      }
}

// ---------------------------------------------------------------------------
// RMSNorm + RoPE, in place on bf16 Q and K laid out [B,H,S,HD].
// One wave per (b,h,s) row; 4 elements per lane; shuffle reductions.
// ---------------------------------------------------------------------------
__global__ __launch_bounds__(256)
void norm_rope(__bf16* __restrict__ Qb, __bf16* __restrict__ Kb,
               const float* __restrict__ qw, const float* __restrict__ kw,
               const float* __restrict__ cosb, const float* __restrict__ sinb) {
  int row  = blockIdx.x * 8 + (threadIdx.x >> 5);   // over B*H*S
  int lane = threadIdx.x & 31;
  int bh = row / S_, s = row % S_, b = bh / H_;
  size_t base = (size_t)row * HD_;
  const float* cp = cosb + ((size_t)b * S_ + s) * HD_;
  const float* sp = sinb + ((size_t)b * S_ + s) * HD_;

#pragma unroll
  for (int which = 0; which < 2; ++which) {
    __bf16* P = which ? Kb : Qb;
    const float* w = which ? kw : qw;
    float x[4];
#pragma unroll
    for (int i = 0; i < 4; ++i) x[i] = (float)P[base + lane * 4 + i];
    float ss = x[0]*x[0] + x[1]*x[1] + x[2]*x[2] + x[3]*x[3];
#pragma unroll
    for (int m = 16; m >= 1; m >>= 1) ss += __shfl_xor(ss, m, 32);
    float inv = rsqrtf(ss * (1.0f / HD_) + 1e-6f);
    float xn[4];
#pragma unroll
    for (int i = 0; i < 4; ++i) xn[i] = x[i] * inv * w[lane * 4 + i];
    // rotate_half partner lives at d +/- 64 -> lane ^ 16, same sub-index
#pragma unroll
    for (int i = 0; i < 4; ++i) {
      float pr  = __shfl_xor(xn[i], 16, 32);
      float rot = (lane < 16) ? -pr : pr;
      int d = lane * 4 + i;
      P[base + d] = (__bf16)(xn[i] * cp[d] + rot * sp[d]);
    }
  }
}

// ---------------------------------------------------------------------------
// Flash attention, causal.  One block per (b,h, 128-row q tile); 8 waves,
// each wave owns 16 q rows.  Q fragments live in registers, K/V tiles in
// LDS, P restaged through per-wave LDS for the PV matmul.
// ---------------------------------------------------------------------------
#define QT 128
#define KT 64

__global__ __launch_bounds__(256)
void flash_attn(const __bf16* __restrict__ Qb, const __bf16* __restrict__ Kb,
                const __bf16* __restrict__ Vb, __bf16* __restrict__ Ob,
                float scale) {
  __shared__ __bf16 Ks[KT][HD_ + 8];
  __shared__ __bf16 Vs[KT][HD_ + 8];
  __shared__ __bf16 Ps[8][16][KT + 8];

  const int bh = blockIdx.y, b = bh / H_, h = bh % H_;
  const int q0 = blockIdx.x * QT;
  const int tid = threadIdx.x, wid = tid >> 5, lane = tid & 31;
  const int half = lane >> 4, l16 = lane & 15;

  const __bf16* Qp = Qb + (size_t)bh * S_ * HD_;
  const __bf16* Kp = Kb + (size_t)bh * S_ * HD_;
  const __bf16* Vp = Vb + (size_t)bh * S_ * HD_;

  // Q A-fragments for this wave's 16 rows, all 4 k-steps (HD = 4 * 32)
  v16bf qfr[4];
  {
    int qrow = q0 + wid * 16 + l16;           // A layout: lane = M
#pragma unroll
    for (int ks = 0; ks < 4; ++ks)
#pragma unroll
      for (int e = 0; e < 16; ++e) {
        int d = ks * 32 + half * 16 + e;
        qfr[ks][e] = Qp[(size_t)qrow * HD_ + d];
      }
  }

  v8f oacc[8] = {};                           // 16 rows x 128 dims, f32
  float mrow[8], lrow[8];
#pragma unroll
  for (int r = 0; r < 8; ++r) { mrow[r] = -1e30f; lrow[r] = 0.0f; }

  const int nkb = (q0 + QT + KT - 1) / KT;    // causal extent for this block
  for (int kb = 0; kb < nkb; ++kb) {
    const int kbase = kb * KT;
    __syncthreads();
    {   // stage K and V tiles: 64x128 bf16 each, 32 elems per thread
      int r = tid >> 2, c = (tid & 3) * 32;
      const __bf16* ksrc = Kp + (size_t)(kbase + r) * HD_ + c;
      const __bf16* vsrc = Vp + (size_t)(kbase + r) * HD_ + c;
#pragma unroll
      for (int i = 0; i < 32; ++i) { Ks[r][c + i] = ksrc[i]; Vs[r][c + i] = vsrc[i]; }
    }
    __syncthreads();

    // scores: 16 rows x 64 keys = 4 N-tiles, K over HD in 4 steps of 32
    v8f sacc[4] = {};
#pragma unroll
    for (int ks = 0; ks < 4; ++ks)
#pragma unroll
      for (int nt = 0; nt < 4; ++nt) {
        v16bf bfr;
        int nr = nt * 16 + l16;
#pragma unroll
        for (int e = 0; e < 16; ++e)
          bfr[e] = Ks[nr][ks * 32 + half * 16 + e];
        sacc[nt] = wmma_bf16(qfr[ks], bfr, sacc[nt]);
      }

    // scale, causal mask, online softmax (16-lane shuffle reductions)
    float alpha[8];
#pragma unroll
    for (int r = 0; r < 8; ++r) {
      int qg = q0 + wid * 16 + r + 8 * half;
      float mx = -3e38f;
#pragma unroll
      for (int nt = 0; nt < 4; ++nt) {
        int kg = kbase + nt * 16 + l16;
        float v = sacc[nt][r] * scale;
        if (kg > qg) v = -3e38f;
        sacc[nt][r] = v;
        mx = fmaxf(mx, v);
      }
#pragma unroll
      for (int m = 8; m >= 1; m >>= 1) mx = fmaxf(mx, __shfl_xor(mx, m, 16));
      float mnew = fmaxf(mrow[r], mx);        // stays >= -1e30 -> exp underflows to 0 for masked
      alpha[r] = __expf(mrow[r] - mnew);
      mrow[r] = mnew;
      float sum = 0.0f;
#pragma unroll
      for (int nt = 0; nt < 4; ++nt) {
        float p = __expf(sacc[nt][r] - mnew);
        sacc[nt][r] = p;
        sum += p;
      }
#pragma unroll
      for (int m = 8; m >= 1; m >>= 1) sum += __shfl_xor(sum, m, 16);
      lrow[r] = lrow[r] * alpha[r] + sum;
    }

    // rescale running O
#pragma unroll
    for (int nt = 0; nt < 8; ++nt)
#pragma unroll
      for (int r = 0; r < 8; ++r) oacc[nt][r] *= alpha[r];

    // restage P (C layout) into per-wave LDS as bf16
#pragma unroll
    for (int nt = 0; nt < 4; ++nt)
#pragma unroll
      for (int r = 0; r < 8; ++r)
        Ps[wid][r + 8 * half][nt * 16 + l16] = (__bf16)sacc[nt][r];

    // PV: P(16x64) * V(64x128); 2 K-steps of 32 keys, 8 N-tiles of dims
#pragma unroll
    for (int ks2 = 0; ks2 < 2; ++ks2) {
      v16bf pfr;
#pragma unroll
      for (int e = 0; e < 16; ++e) {
        int kk = ks2 * 32 + half * 16 + e;
        pfr[e] = Ps[wid][l16][kk];
      }
#pragma unroll
      for (int nt = 0; nt < 8; ++nt) {
        v16bf bfr;
#pragma unroll
        for (int e = 0; e < 16; ++e)
          bfr[e] = Vs[ks2 * 32 + half * 16 + e][nt * 16 + l16];
        oacc[nt] = wmma_bf16(pfr, bfr, oacc[nt]);
      }
    }
  }

  // epilogue: O / l, write bf16 to [B,S,H*HD] for the output projection
#pragma unroll
  for (int nt = 0; nt < 8; ++nt)
#pragma unroll
    for (int r = 0; r < 8; ++r) {
      int qg = q0 + wid * 16 + r + 8 * half;
      int d  = nt * 16 + l16;
      float v = oacc[nt][r] / lrow[r];
      Ob[((size_t)b * S_ + qg) * (H_ * HD_) + h * HD_ + d] = (__bf16)v;
    }
}

// ---------------------------------------------------------------------------
// Host-side orchestration
// ---------------------------------------------------------------------------
extern "C" void kernel_launch(void* const* d_in, const int* in_sizes, int n_in,
                              void* d_out, int out_size, void* d_ws, size_t ws_size,
                              hipStream_t stream) {
  const float* X    = (const float*)d_in[0];   // hidden_states [B,S,D]
  const float* cosb = (const float*)d_in[1];
  const float* sinb = (const float*)d_in[2];
  // d_in[3] = attention_mask (causal; computed analytically in-kernel)
  const float* Wq = (const float*)d_in[4];
  const float* bq = (const float*)d_in[5];
  const float* Wk = (const float*)d_in[6];
  const float* bk = (const float*)d_in[7];
  const float* Wv = (const float*)d_in[8];
  const float* bv = (const float*)d_in[9];
  const float* Wo = (const float*)d_in[10];
  const float* qnw = (const float*)d_in[11];
  const float* knw = (const float*)d_in[12];

  // workspace layout (bf16 buffers)
  char* ws = (char*)d_ws;
  const size_t szX = (size_t)MTOT * D_ * 2;          // 16 MB
  const size_t szW = (size_t)D_ * D_ * 2;            //  8 MB each
  const size_t szA = (size_t)MTOT * D_ * 2;          // 16 MB each
  __bf16* Xb  = (__bf16*)(ws);
  __bf16* Wqb = (__bf16*)(ws + szX);
  __bf16* Wkb = (__bf16*)(ws + szX + szW);
  __bf16* Wvb = (__bf16*)(ws + szX + 2 * szW);
  __bf16* Wob = (__bf16*)(ws + szX + 3 * szW);
  __bf16* Qbuf = (__bf16*)(ws + szX + 4 * szW);              // [B,H,S,HD]
  __bf16* Kbuf = (__bf16*)(ws + szX + 4 * szW + szA);
  __bf16* Vbuf = (__bf16*)(ws + szX + 4 * szW + 2 * szA);
  __bf16* Attn = (__bf16*)(ws + szX + 4 * szW + 3 * szA);    // [B,S,H*HD]

  // 1) f32 -> bf16
  {
    long nx = (long)MTOT * D_, nw = (long)D_ * D_;
    cvt_f32_bf16<<<(nx + 255) / 256, 256, 0, stream>>>(X,  Xb,  nx);
    cvt_f32_bf16<<<(nw + 255) / 256, 256, 0, stream>>>(Wq, Wqb, nw);
    cvt_f32_bf16<<<(nw + 255) / 256, 256, 0, stream>>>(Wk, Wkb, nw);
    cvt_f32_bf16<<<(nw + 255) / 256, 256, 0, stream>>>(Wv, Wvb, nw);
    cvt_f32_bf16<<<(nw + 255) / 256, 256, 0, stream>>>(Wo, Wob, nw);
  }

  // 2) Q/K/V projections (bf16 WMMA, epilogue transposes to [B,H,S,HD])
  dim3 ggrid(D_ / TN, MTOT / TM);   // 32 x 32
  gemm_bf16_wmma<<<ggrid, 256, 0, stream>>>(Xb, Wqb, bq, Qbuf, MTOT, D_, D_, 0);
  gemm_bf16_wmma<<<ggrid, 256, 0, stream>>>(Xb, Wkb, bk, Kbuf, MTOT, D_, D_, 0);
  gemm_bf16_wmma<<<ggrid, 256, 0, stream>>>(Xb, Wvb, bv, Vbuf, MTOT, D_, D_, 0);

  // 3) RMSNorm + RoPE on Q, K (in place)
  norm_rope<<<(B_ * H_ * S_) / 8, 256, 0, stream>>>(Qbuf, Kbuf, qnw, knw, cosb, sinb);

  // 4) causal flash attention
  {
    dim3 fgrid(S_ / QT, B_ * H_);   // 16 x 32
    float scale = 1.0f / sqrtf((float)HD_);
    flash_attn<<<fgrid, 256, 0, stream>>>(Qbuf, Kbuf, Vbuf, Attn, scale);
  }

  // 5) output projection -> f32 d_out
  gemm_bf16_wmma<<<ggrid, 256, 0, stream>>>(Attn, Wob, nullptr, (float*)d_out,
                                            MTOT, D_, D_, 1);
}
